// MoELayer_77472620085636
// MI455X (gfx1250) — matrix operand order repaired and checked by reading, not compiled, not run on hardware
//
#include <hip/hip_runtime.h>

#define HD 4096      // hidden size
#define ID 5632      // intermediate size
#define NE 8         // experts
#define NT 4096      // tokens (2*2048)
#define NK 2         // top-k
#define LDP 20       // LDS row stride in dwords: 80B = 16B-aligned rows, conflict-free (gcd trick)

typedef __attribute__((ext_vector_type(16))) __bf16 v16bf;
typedef __attribute__((ext_vector_type(8)))  float  v8f;
typedef __attribute__((ext_vector_type(4)))  int    v4i;

union Frag16 { v16bf v; unsigned u[8]; };

#if defined(__AMDGCN__) && __has_builtin(__builtin_amdgcn_cvt_pk_bf16_f32)
typedef __attribute__((ext_vector_type(2))) __bf16 v2bf;
static __device__ __forceinline__ unsigned packbf(float a, float b) {
    v2bf r = __builtin_amdgcn_cvt_pk_bf16_f32(a, b);
    return __builtin_bit_cast(unsigned, r);
}
#else
static __device__ __forceinline__ unsigned short f2bf_(float f) {
    unsigned u = __builtin_bit_cast(unsigned, f);
    u += 0x7FFFu + ((u >> 16) & 1u);           // round-to-nearest-even
    return (unsigned short)(u >> 16);
}
static __device__ __forceinline__ unsigned packbf(float a, float b) {
    return (unsigned)f2bf_(a) | ((unsigned)f2bf_(b) << 16);
}
#endif

static __device__ __forceinline__ unsigned short f2bf(float f) {
    unsigned u = __builtin_bit_cast(unsigned, f);
    u += 0x7FFFu + ((u >> 16) & 1u);
    return (unsigned short)(u >> 16);
}

#if defined(__gfx1250__) && __has_builtin(__builtin_amdgcn_global_load_async_to_lds_b128) && __has_builtin(__builtin_amdgcn_s_wait_asynccnt)
#define USE_ASYNC_LDS 1
#else
#define USE_ASYNC_LDS 0
#endif

static __device__ __forceinline__ v8f vzero8() {
    v8f z = {0.f,0.f,0.f,0.f,0.f,0.f,0.f,0.f};
    return z;
}

// ---------------- init: zero output + counters ----------------
__global__ void zero_kernel(float* __restrict__ out, int* cnt, float* pm, float* zs) {
    size_t n = (size_t)NT * HD;
    size_t idx = (size_t)blockIdx.x * blockDim.x + threadIdx.x;
    size_t stride = (size_t)gridDim.x * blockDim.x;
    for (size_t i = idx; i < n; i += stride) out[i] = 0.f;
    if (blockIdx.x == 0) {
        if (threadIdx.x < NE) { cnt[threadIdx.x] = 0; pm[threadIdx.x] = 0.f; }
        if (threadIdx.x == NE) *zs = 0.f;
    }
}

// ---------------- router: logits, softmax, top-2, aux stats ----------------
__global__ __launch_bounds__(256)
void router_kernel(const float* __restrict__ x, const float* __restrict__ Wgate,
                   int* __restrict__ cnt, int* __restrict__ rowmap,
                   float* __restrict__ roww, float* __restrict__ pm, float* __restrict__ zs) {
    int lane = threadIdx.x & 31;
    int wid  = threadIdx.x >> 5;
    int t = blockIdx.x * 8 + wid;              // one wave per token
    float acc[NE];
    #pragma unroll
    for (int e = 0; e < NE; ++e) acc[e] = 0.f;
    const float* xr = x + (size_t)t * HD;
    for (int j = lane; j < HD; j += 32) {
        float xv = xr[j];
        #pragma unroll
        for (int e = 0; e < NE; ++e) acc[e] += xv * Wgate[e * HD + j];
    }
    #pragma unroll
    for (int off = 16; off > 0; off >>= 1) {
        #pragma unroll
        for (int e = 0; e < NE; ++e) acc[e] += __shfl_xor(acc[e], off, 32);
    }
    if (lane == 0) {
        float mx = acc[0];
        #pragma unroll
        for (int e = 1; e < NE; ++e) mx = fmaxf(mx, acc[e]);
        float p[NE]; float sum = 0.f;
        #pragma unroll
        for (int e = 0; e < NE; ++e) { p[e] = __expf(acc[e] - mx); sum += p[e]; }
        float inv = 1.0f / sum;
        #pragma unroll
        for (int e = 0; e < NE; ++e) { p[e] *= inv; atomicAdd(&pm[e], p[e]); }
        float lse = __logf(sum) + mx;
        atomicAdd(zs, lse * lse);
        int i0 = 0;
        #pragma unroll
        for (int e = 1; e < NE; ++e) if (p[e] > p[i0]) i0 = e;
        int i1 = (i0 == 0) ? 1 : 0;
        #pragma unroll
        for (int e = 0; e < NE; ++e) if (e != i0 && p[e] > p[i1]) i1 = e;
        float w0 = p[i0], w1 = p[i1];
        float s2 = w0 + w1;
        w0 /= s2; w1 /= s2;
        int pos0 = atomicAdd(&cnt[i0], 1);
        rowmap[i0 * NT + pos0] = t; roww[i0 * NT + pos0] = w0;
        int pos1 = atomicAdd(&cnt[i1], 1);
        rowmap[i1 * NT + pos1] = t; roww[i1 * NT + pos1] = w1;
    }
}

// ---------------- aux loss finalize ----------------
__global__ void aux_kernel(const int* __restrict__ cnt, const float* __restrict__ pm,
                           const float* __restrict__ zs, float* __restrict__ auxout) {
    float s = 0.f;
    for (int e = 0; e < NE; ++e)
        s += (pm[e] / (float)NT) * ((float)cnt[e] / (float)(NT * NK));
    float aux = (float)NE * s;
    float z = *zs / (float)NT;
    *auxout = 0.02f * aux + 0.001f * z;
}

// ---------------- fused gate/up: act = silu(x@WgT) * (x@WuT)  (bf16 out) ----------------
__global__ __launch_bounds__(256)
void moe_gateup_kernel(const float* __restrict__ x,
                       const float* __restrict__ Wg_e,
                       const float* __restrict__ Wu_e,
                       const int* __restrict__ cntp,
                       const int* __restrict__ rowmap_e,
                       unsigned short* __restrict__ act) {
    __shared__ unsigned As[128][LDP];
    __shared__ unsigned Bg[64][LDP];
    __shared__ unsigned Bu[64][LDP];

    int cnt = *cntp;
    int m0 = blockIdx.y * 128;
    if (m0 >= cnt) return;                     // early exit past routed rows
    int n0 = blockIdx.x * 64;

    int tid  = threadIdx.x;
    int lane = tid & 31;
    int lm   = lane & 15;
    int hh   = lane >> 4;
    int wid  = tid >> 5;
    int waveM = wid & 3;                       // 4 wave-rows of 32
    int waveN = wid >> 2;                      // 2 wave-cols of 32

    v8f accG[2][2], accU[2][2];
    #pragma unroll
    for (int a = 0; a < 2; ++a)
        #pragma unroll
        for (int b = 0; b < 2; ++b) { accG[a][b] = vzero8(); accU[a][b] = vzero8(); }

    for (int k0 = 0; k0 < HD; k0 += 32) {
        // stage A (gathered token rows), fp32 -> bf16 pairs
        #pragma unroll
        for (int it = 0; it < 4; ++it) {
            int idx = tid + it * 256;          // 1024 float4 chunks
            int r = idx >> 3, c = idx & 7;
            int rg = m0 + r;
            int tok = (rg < cnt) ? rowmap_e[rg] : rowmap_e[m0];
            float4 f = reinterpret_cast<const float4*>(x + (size_t)tok * HD + k0)[c];
            As[r][c * 2]     = packbf(f.x, f.y);
            As[r][c * 2 + 1] = packbf(f.z, f.w);
        }
        // stage Wg/Wu tiles
        #pragma unroll
        for (int it = 0; it < 2; ++it) {
            int idx = tid + it * 256;          // 512 chunks per matrix
            int r = idx >> 3, c = idx & 7;
            const float* gr = Wg_e + (size_t)(n0 + r) * HD + k0;
            const float* ur = Wu_e + (size_t)(n0 + r) * HD + k0;
            if (k0 + 32 < HD) { __builtin_prefetch(gr + 32, 0, 1); __builtin_prefetch(ur + 32, 0, 1); }
            float4 f = reinterpret_cast<const float4*>(gr)[c];
            Bg[r][c*2]   = packbf(f.x, f.y);
            Bg[r][c*2+1] = packbf(f.z, f.w);
            float4 g2 = reinterpret_cast<const float4*>(ur)[c];
            Bu[r][c*2]   = packbf(g2.x, g2.y);
            Bu[r][c*2+1] = packbf(g2.z, g2.w);
        }
        __syncthreads();

        Frag16 a[2], bg[2], bu[2];
        #pragma unroll
        for (int mi = 0; mi < 2; ++mi) {
            int m = waveM * 32 + mi * 16 + lm;
            #pragma unroll
            for (int i = 0; i < 8; ++i) {
                int kk = (i < 4 ? i : i + 4) + hh * 4;   // CDNA5 A(16x32 bf16) layout
                a[mi].u[i] = As[m][kk];
            }
        }
        #pragma unroll
        for (int ni = 0; ni < 2; ++ni) {
            int n = waveN * 32 + ni * 16 + lm;
            #pragma unroll
            for (int j = 0; j < 8; ++j) {                // CDNA5 B(32x16 bf16) layout
                bg[ni].u[j] = Bg[n][(j < 4 ? j : j + 4) + hh * 4];
                bu[ni].u[j] = Bu[n][(j < 4 ? j : j + 4) + hh * 4];
            }
        }
        #pragma unroll
        for (int mi = 0; mi < 2; ++mi)
            #pragma unroll
            for (int ni = 0; ni < 2; ++ni) {
                accG[mi][ni] = __builtin_amdgcn_wmma_f32_16x16x32_bf16(
                    false, a[mi].v, false, bg[ni].v, (short)0, accG[mi][ni], false, false);
                accU[mi][ni] = __builtin_amdgcn_wmma_f32_16x16x32_bf16(
                    false, a[mi].v, false, bu[ni].v, (short)0, accU[mi][ni], false, false);
            }
        __syncthreads();
    }

    // epilogue: silu(g)*u -> bf16 activation buffer
    #pragma unroll
    for (int mi = 0; mi < 2; ++mi) {
        #pragma unroll
        for (int i = 0; i < 8; ++i) {
            int ml = waveM * 32 + mi * 16 + i + 8 * hh;  // C frag: M = i + 8*(lane/16)
            int row = m0 + ml;
            if (row < cnt) {
                #pragma unroll
                for (int ni = 0; ni < 2; ++ni) {
                    int n = n0 + waveN * 32 + ni * 16 + lm;
                    float g = accG[mi][ni][i];
                    float u = accU[mi][ni][i];
                    float s = g / (1.0f + __expf(-g));
                    act[(size_t)row * ID + n] = f2bf(s * u);
                }
            }
        }
    }
}

// ---------------- down proj: out[tok] += w * (act @ WdT) ----------------
__global__ __launch_bounds__(256)
void moe_down_kernel(const unsigned short* __restrict__ act,
                     const float* __restrict__ Wd_e,
                     const int* __restrict__ cntp,
                     const int* __restrict__ rowmap_e,
                     const float* __restrict__ roww_e,
                     float* __restrict__ out) {
    __shared__ unsigned As[128][LDP];
    __shared__ unsigned Bs[128][LDP];

    int cnt = *cntp;
    int m0 = blockIdx.y * 128;
    if (m0 >= cnt) return;
    int n0 = blockIdx.x * 128;

    int tid  = threadIdx.x;
    int lane = tid & 31;
    int lm   = lane & 15;
    int hh   = lane >> 4;
    int wid  = tid >> 5;
    int waveM = wid & 3;                       // 4 x 32 rows
    int waveN = wid >> 2;                      // 2 x 64 cols

    v8f acc[2][4];
    #pragma unroll
    for (int a = 0; a < 2; ++a)
        #pragma unroll
        for (int b = 0; b < 4; ++b) acc[a][b] = vzero8();

    for (int k0 = 0; k0 < ID; k0 += 32) {
        // A: bf16 activations (already packed pairs)
#if USE_ASYNC_LDS
        // HBM -> LDS DMA, no VGPR roundtrip; tracked by ASYNCcnt
        #pragma unroll
        for (int it = 0; it < 2; ++it) {
            int idx = tid + it * 256;          // 512 b128 chunks
            int r = idx >> 2, c = idx & 3;
            int rg = m0 + r;
            int rr = (rg < cnt) ? rg : m0;
            const unsigned short* gp = act + (size_t)rr * ID + k0 + c * 8;
            __builtin_amdgcn_global_load_async_to_lds_b128(
                (__attribute__((address_space(1))) v4i*)((void*)gp),
                (__attribute__((address_space(3))) v4i*)((void*)&As[r][c * 4]), 0, 0);
        }
#else
        #pragma unroll
        for (int it = 0; it < 2; ++it) {
            int idx = tid + it * 256;          // 512 uint4 chunks
            int r = idx >> 2, c = idx & 3;
            int rg = m0 + r;
            int rr = (rg < cnt) ? rg : m0;
            uint4 q = reinterpret_cast<const uint4*>(act + (size_t)rr * ID + k0)[c];
            As[r][c*4+0] = q.x; As[r][c*4+1] = q.y;
            As[r][c*4+2] = q.z; As[r][c*4+3] = q.w;
        }
#endif
        // B: Wd rows fp32 -> bf16
        #pragma unroll
        for (int it = 0; it < 4; ++it) {
            int idx = tid + it * 256;          // 1024 float4 chunks
            int r = idx >> 3, c = idx & 7;
            const float* wr = Wd_e + (size_t)(n0 + r) * ID + k0;
            if (k0 + 32 < ID) __builtin_prefetch(wr + 32, 0, 1);
            float4 f = reinterpret_cast<const float4*>(wr)[c];
            Bs[r][c*2]   = packbf(f.x, f.y);
            Bs[r][c*2+1] = packbf(f.z, f.w);
        }
#if USE_ASYNC_LDS
        __builtin_amdgcn_s_wait_asynccnt(0);
#endif
        __syncthreads();

        Frag16 a[2], b[4];
        #pragma unroll
        for (int mi = 0; mi < 2; ++mi) {
            int m = waveM * 32 + mi * 16 + lm;
            #pragma unroll
            for (int i = 0; i < 8; ++i) {
                int kk = (i < 4 ? i : i + 4) + hh * 4;
                a[mi].u[i] = As[m][kk];
            }
        }
        #pragma unroll
        for (int ni = 0; ni < 4; ++ni) {
            int n = waveN * 64 + ni * 16 + lm;
            #pragma unroll
            for (int j = 0; j < 8; ++j) b[ni].u[j] = Bs[n][(j < 4 ? j : j + 4) + hh * 4];
        }
        #pragma unroll
        for (int mi = 0; mi < 2; ++mi)
            #pragma unroll
            for (int ni = 0; ni < 4; ++ni)
                acc[mi][ni] = __builtin_amdgcn_wmma_f32_16x16x32_bf16(
                    false, a[mi].v, false, b[ni].v, (short)0, acc[mi][ni], false, false);
        __syncthreads();
    }

    // epilogue: weighted scatter-add into out (each token touched by exactly 2 experts)
    #pragma unroll
    for (int mi = 0; mi < 2; ++mi) {
        #pragma unroll
        for (int i = 0; i < 8; ++i) {
            int ml = waveM * 32 + mi * 16 + i + 8 * hh;
            int row = m0 + ml;
            if (row < cnt) {
                int tok = rowmap_e[row];
                float w = roww_e[row];
                #pragma unroll
                for (int ni = 0; ni < 4; ++ni) {
                    int n = n0 + waveN * 64 + ni * 16 + lm;
                    atomicAdd(&out[(size_t)tok * HD + n], w * acc[mi][ni][i]);
                }
            }
        }
    }
}

extern "C" void kernel_launch(void* const* d_in, const int* in_sizes, int n_in,
                              void* d_out, int out_size, void* d_ws, size_t ws_size,
                              hipStream_t stream) {
    (void)in_sizes; (void)n_in; (void)out_size; (void)ws_size;
    const float* x     = (const float*)d_in[0];
    const float* Wgate = (const float*)d_in[1];
    const float* Wg    = (const float*)d_in[2];
    const float* Wu    = (const float*)d_in[3];
    const float* Wd    = (const float*)d_in[4];
    float* out = (float*)d_out;

    char* w = (char*)d_ws;
    int*   cnt    = (int*)w;                              // NE ints
    float* pm     = (float*)(w + 32);                     // NE floats (softmax mean accum)
    float* zs     = (float*)(w + 64);                     // 1 float (z-loss accum)
    int*   rowmap = (int*)(w + 128);                      // NE*NT ints
    float* roww   = (float*)(w + 128 + 4 * NE * NT);      // NE*NT floats
    unsigned short* act = (unsigned short*)(w + 262400);  // NT*ID bf16 (~46 MB), 256B aligned

    zero_kernel<<<2048, 256, 0, stream>>>(out, cnt, pm, zs);
    router_kernel<<<NT / 8, 256, 0, stream>>>(x, Wgate, cnt, rowmap, roww, pm, zs);
    aux_kernel<<<1, 1, 0, stream>>>(cnt, pm, zs, out + (size_t)NT * HD);

    for (int e = 0; e < NE; ++e) {
        moe_gateup_kernel<<<dim3(ID / 64, NT / 128), 256, 0, stream>>>(
            x, Wg + (size_t)e * ID * HD, Wu + (size_t)e * ID * HD,
            cnt + e, rowmap + e * NT, act);
        moe_down_kernel<<<dim3(HD / 128, NT / 128), 256, 0, stream>>>(
            act, Wd + (size_t)e * HD * ID,
            cnt + e, rowmap + e * NT, roww + e * NT, out);
    }
}
